// GPT_14310831030814
// MI455X (gfx1250) — matrix operand order repaired and checked by reading, not compiled, not run on hardware
//
#include <hip/hip_runtime.h>
#include <hip/hip_bf16.h>

// ---- problem constants (match reference) ----
#define BB   4
#define TT   512
#define DD   512
#define HH   8
#define LL   6
#define HDIM 64            // DD/HH
#define VV   50257
#define VPAD 50304         // VV padded to multiple of 128
#define MTOT (BB*TT)       // 2048 rows
#define FF   (4*DD)        // 2048

typedef __bf16 bf16_t;
typedef bf16_t v16bf __attribute__((ext_vector_type(16)));
typedef bf16_t v8bf  __attribute__((ext_vector_type(8)));
typedef float  v8f   __attribute__((ext_vector_type(8)));

union ABfrag { v16bf v; v8bf h[2]; };

// ---------------------------------------------------------------------------
// WMMA GEMM with 2x{NT} register blocking per wave (each tile 16x16):
//   C[m,n] = sum_k A[m,k]*Bt[n,k]  (+bias[n]) (gelu) (+resid)  -> f32 or bf16
//   A  : bf16 [M,K] row-major (K contiguous)
//   Bt : bf16 [Npad,K] row-major (K contiguous)  == B transposed
// Block = 256 threads = 8 waves (2x4 wave grid); block covers 64 x (NT*64).
// grid = (Npad/(NT*64), M/64, Z).
// Per-z C offset: cz = (z/zdiv)*sCz1 + (z%zdiv)*sCz2   (zdiv=1 -> plain batch;
// zdiv=HH -> scatter attention heads straight into merged [B,T,D] layout).
// ---------------------------------------------------------------------------
template<int NT, bool GELU, bool CBF16>
__global__ __launch_bounds__(256) void gemm_wmma_bf16(
    const bf16_t* __restrict__ A, const bf16_t* __restrict__ Bt,
    const float* __restrict__ bias, const float* resid, void* Cout,
    int K, int Nvalid, int ldc,
    long long sAz, long long sBz, int zdiv, long long sCz1, long long sCz2)
{
  const int z = blockIdx.z;
  A  += (long long)z * sAz;
  Bt += (long long)z * sBz;
  const long long cz = (long long)(z / zdiv) * sCz1 + (long long)(z % zdiv) * sCz2;

  const int wave = threadIdx.x >> 5;
  const int lane = threadIdx.x & 31;
  const int wm = wave >> 2;          // 0..1
  const int wn = wave & 3;           // 0..3
  const int m0 = blockIdx.y * 64 + wm * 32;           // wave: rows m0..m0+31
  const int n0 = blockIdx.x * (NT * 64) + wn * (NT * 16);
  const int lr = lane & 15;
  const int hi = lane >> 4;

  // A fragment: lane holds row; K chunks hi*8+{0..7} and 16+hi*8+{0..7}
  const bf16_t* ap0 = A + (long long)(m0 + lr) * K + hi * 8;
  const bf16_t* ap1 = ap0 + 16LL * K;
  // B fragment: lane holds col; K chunks hi*8+{0..7} and 16+hi*8+{0..7}
  const bf16_t* bp0 = Bt + (long long)(n0 + lr) * K + hi * 8;
  const bf16_t* bp1 = bp0 + 16LL * K;

  v8f acc[2][NT];
#pragma unroll
  for (int i = 0; i < 2; ++i)
#pragma unroll
    for (int j = 0; j < NT; ++j)
      acc[i][j] = (v8f){0.f,0.f,0.f,0.f,0.f,0.f,0.f,0.f};

  for (int k = 0; k < K; k += 32) {
    ABfrag a0, a1, b0, b1;
    a0.h[0] = *(const v8bf*)(ap0 + k);
    a0.h[1] = *(const v8bf*)(ap0 + k + 16);
    a1.h[0] = *(const v8bf*)(ap1 + k);
    a1.h[1] = *(const v8bf*)(ap1 + k + 16);
    b0.h[0] = *(const v8bf*)(bp0 + k);
    b0.h[1] = *(const v8bf*)(bp0 + k + 16);
    if constexpr (NT == 2) {
      b1.h[0] = *(const v8bf*)(bp1 + k);
      b1.h[1] = *(const v8bf*)(bp1 + k + 16);
    }
    acc[0][0] = __builtin_amdgcn_wmma_f32_16x16x32_bf16(
        false, a0.v, false, b0.v, (short)0, acc[0][0], false, false);
    acc[1][0] = __builtin_amdgcn_wmma_f32_16x16x32_bf16(
        false, a1.v, false, b0.v, (short)0, acc[1][0], false, false);
    if constexpr (NT == 2) {
      acc[0][1] = __builtin_amdgcn_wmma_f32_16x16x32_bf16(
          false, a0.v, false, b1.v, (short)0, acc[0][1], false, false);
      acc[1][1] = __builtin_amdgcn_wmma_f32_16x16x32_bf16(
          false, a1.v, false, b1.v, (short)0, acc[1][1], false, false);
    }
  }

  // C layout per tile: lanes 0-15 -> N=lane, VGPR r -> M=r; lanes 16-31 -> M=r+8
#pragma unroll
  for (int j = 0; j < NT; ++j) {
    const int n = n0 + j * 16 + lr;
    if (n >= Nvalid) continue;
    const float bv = bias ? bias[n] : 0.0f;
#pragma unroll
    for (int i = 0; i < 2; ++i) {
#pragma unroll
      for (int r = 0; r < 8; ++r) {
        const int m = m0 + i * 16 + hi * 8 + r;
        float v = acc[i][j][r] + bv;
        if (GELU) v = 0.5f * v * (1.0f + erff(v * 0.70710678118654752f));
        const long long ci = cz + (long long)m * ldc + n;
        if constexpr (CBF16) {
          ((bf16_t*)Cout)[ci] = (bf16_t)v;
        } else {
          if (resid) v += resid[ci];
          ((float*)Cout)[ci] = v;
        }
      }
    }
  }
}

// ---- transpose-convert weight f32[K,N] -> bf16[Npad,K], zero pad n>=N ----
__global__ void wconv_kernel(const float* __restrict__ src, bf16_t* __restrict__ dst,
                             int K, int N, long long total)
{
  long long idx = (long long)blockIdx.x * 256 + threadIdx.x;
  if (idx >= total) return;
  long long n = idx / K;
  int k = (int)(idx % K);
  dst[idx] = (n < N) ? (bf16_t)src[(long long)k * N + n] : (bf16_t)0.0f;
}

// ---- token + position embedding ----
__global__ void embed_kernel(const int* __restrict__ x, const float* __restrict__ tok,
                             const float* __restrict__ pos, float* __restrict__ h)
{
  int idx = blockIdx.x * 256 + threadIdx.x;
  if (idx >= MTOT * DD) return;
  int d = idx % DD;
  int bt = idx / DD;
  int t = bt % TT;
  h[idx] = tok[(long long)x[bt] * DD + d] + pos[t * DD + d];
}

// ---- LayerNorm over D=512, output bf16 (feeds GEMM A operand) ----
__global__ __launch_bounds__(256) void ln_kernel(const float* __restrict__ h,
    const float* __restrict__ sc, const float* __restrict__ bi,
    bf16_t* __restrict__ out)
{
  __shared__ float red[256];
  const int row = blockIdx.x;
  const float* x = h + (long long)row * DD;
  const int t0 = threadIdx.x, t1 = threadIdx.x + 256;
  const float v0 = x[t0], v1 = x[t1];
  red[threadIdx.x] = v0 + v1; __syncthreads();
  for (int o = 128; o > 0; o >>= 1) {
    if (threadIdx.x < o) red[threadIdx.x] += red[threadIdx.x + o];
    __syncthreads();
  }
  const float mean = red[0] * (1.0f / DD); __syncthreads();
  const float d0 = v0 - mean, d1 = v1 - mean;
  red[threadIdx.x] = d0 * d0 + d1 * d1; __syncthreads();
  for (int o = 128; o > 0; o >>= 1) {
    if (threadIdx.x < o) red[threadIdx.x] += red[threadIdx.x + o];
    __syncthreads();
  }
  const float rstd = rsqrtf(red[0] * (1.0f / DD) + 1e-5f);
  out[(long long)row * DD + t0] = (bf16_t)(d0 * rstd * sc[t0] + bi[t0]);
  out[(long long)row * DD + t1] = (bf16_t)(d1 * rstd * sc[t1] + bi[t1]);
}

// ---- split qkv: q,k -> [B,H,T,HD] bf16 ; v -> transposed [B,H,HD,T] bf16 ----
__global__ void split_qkv_kernel(const float* __restrict__ qkv,
                                 bf16_t* __restrict__ q, bf16_t* __restrict__ k,
                                 bf16_t* __restrict__ vT)
{
  long long idx = (long long)blockIdx.x * 256 + threadIdx.x;   // B*T*3D
  if (idx >= (long long)MTOT * 3 * DD) return;
  const int c = (int)(idx % (3 * DD));
  const long long bt = idx / (3 * DD);
  const int t = (int)(bt % TT);
  const int b = (int)(bt / TT);
  const int part = c / DD;
  const int ci = c % DD;
  const int hh = ci / HDIM;
  const int hd = ci % HDIM;
  const bf16_t val = (bf16_t)qkv[idx];
  const long long zi = (long long)b * HH + hh;
  if (part == 0)      q [(zi * TT + t) * HDIM + hd] = val;
  else if (part == 1) k [(zi * TT + t) * HDIM + hd] = val;
  else                vT[(zi * HDIM + hd) * TT + t] = val;
}

// ---- causal softmax over T=512 with 1/sqrt(HD) scale, output bf16 ----
__global__ __launch_bounds__(256) void softmax_kernel(const float* __restrict__ scores,
                                                      bf16_t* __restrict__ attn)
{
  __shared__ float red[256];
  const int row = blockIdx.x;        // B*H*T rows
  const int q = row % TT;
  const float* s = scores + (long long)row * TT;
  bf16_t* o = attn + (long long)row * TT;
  const int t0 = threadIdx.x, t1 = threadIdx.x + 256;
  const float NEG = -__builtin_inff();
  const float v0 = (t0 <= q) ? s[t0] * 0.125f : NEG;
  const float v1 = (t1 <= q) ? s[t1] * 0.125f : NEG;
  red[threadIdx.x] = fmaxf(v0, v1); __syncthreads();
  for (int o2 = 128; o2 > 0; o2 >>= 1) {
    if (threadIdx.x < o2) red[threadIdx.x] = fmaxf(red[threadIdx.x], red[threadIdx.x + o2]);
    __syncthreads();
  }
  const float mx = red[0]; __syncthreads();
  const float e0 = (t0 <= q) ? expf(v0 - mx) : 0.0f;
  const float e1 = (t1 <= q) ? expf(v1 - mx) : 0.0f;
  red[threadIdx.x] = e0 + e1; __syncthreads();
  for (int o2 = 128; o2 > 0; o2 >>= 1) {
    if (threadIdx.x < o2) red[threadIdx.x] += red[threadIdx.x + o2];
    __syncthreads();
  }
  const float inv = 1.0f / red[0];
  o[t0] = (bf16_t)(e0 * inv);
  o[t1] = (bf16_t)(e1 * inv);
}

// ---------------------------------------------------------------------------
extern "C" void kernel_launch(void* const* d_in, const int* in_sizes, int n_in,
                              void* d_out, int out_size, void* d_ws, size_t ws_size,
                              hipStream_t stream)
{
  const int*   x      = (const int*)  d_in[0];
  const float* tok    = (const float*)d_in[1];
  const float* pos    = (const float*)d_in[2];
  const float* qkv_w  = (const float*)d_in[3];
  const float* qkv_b  = (const float*)d_in[4];
  const float* out_w  = (const float*)d_in[5];
  const float* out_b  = (const float*)d_in[6];
  const float* ln1_s  = (const float*)d_in[7];
  const float* ln1_b  = (const float*)d_in[8];
  const float* ff1_w  = (const float*)d_in[9];
  const float* ff1_b  = (const float*)d_in[10];
  const float* ff2_w  = (const float*)d_in[11];
  const float* ff2_b  = (const float*)d_in[12];
  const float* ln2_s  = (const float*)d_in[13];
  const float* ln2_b  = (const float*)d_in[14];
  const float* lnf_s  = (const float*)d_in[15];
  const float* lnf_b  = (const float*)d_in[16];
  const float* head_w = (const float*)d_in[17];
  const float* head_b = (const float*)d_in[18];
  float* logits = (float*)d_out;

  // carve workspace (256B-aligned chunks)
  char* p = (char*)d_ws;
  auto alloc = [&](size_t bytes) -> void* {
    void* r = (void*)p;
    p += (bytes + 255) & ~(size_t)255;
    return r;
  };
  bf16_t* qkvwt = (bf16_t*)alloc((size_t)LL * 3 * DD * DD * 2);
  bf16_t* outwt = (bf16_t*)alloc((size_t)LL * DD * DD * 2);
  bf16_t* ff1wt = (bf16_t*)alloc((size_t)LL * FF * DD * 2);
  bf16_t* ff2wt = (bf16_t*)alloc((size_t)LL * DD * FF * 2);
  bf16_t* headwt= (bf16_t*)alloc((size_t)VPAD * DD * 2);
  float*  h     = (float*) alloc((size_t)MTOT * DD * 4);
  bf16_t* xb    = (bf16_t*)alloc((size_t)MTOT * DD * 2);
  float*  qkvf  = (float*) alloc((size_t)MTOT * 3 * DD * 4);
  bf16_t* qb    = (bf16_t*)alloc((size_t)MTOT * DD * 2);
  bf16_t* kb    = (bf16_t*)alloc((size_t)MTOT * DD * 2);
  bf16_t* vT    = (bf16_t*)alloc((size_t)MTOT * DD * 2);
  float*  scores= (float*) alloc((size_t)BB * HH * TT * TT * 4);
  bf16_t* attn  = (bf16_t*)alloc((size_t)BB * HH * TT * TT * 2);
  bf16_t* obf   = (bf16_t*)alloc((size_t)MTOT * DD * 2);
  bf16_t* ffb   = (bf16_t*)alloc((size_t)MTOT * FF * 2);

  auto blks = [](long long n) { return (unsigned)((n + 255) / 256); };

  // ---- weight conversion (transpose to [N,K] bf16) ----
  for (int l = 0; l < LL; ++l) {
    wconv_kernel<<<blks((long long)3 * DD * DD), 256, 0, stream>>>(
        qkv_w + (size_t)l * DD * 3 * DD, qkvwt + (size_t)l * 3 * DD * DD,
        DD, 3 * DD, (long long)3 * DD * DD);
    wconv_kernel<<<blks((long long)DD * DD), 256, 0, stream>>>(
        out_w + (size_t)l * DD * DD, outwt + (size_t)l * DD * DD,
        DD, DD, (long long)DD * DD);
    wconv_kernel<<<blks((long long)FF * DD), 256, 0, stream>>>(
        ff1_w + (size_t)l * DD * FF, ff1wt + (size_t)l * FF * DD,
        DD, FF, (long long)FF * DD);
    wconv_kernel<<<blks((long long)DD * FF), 256, 0, stream>>>(
        ff2_w + (size_t)l * FF * DD, ff2wt + (size_t)l * DD * FF,
        FF, DD, (long long)DD * FF);
  }
  wconv_kernel<<<blks((long long)VPAD * DD), 256, 0, stream>>>(
      head_w, headwt, DD, VV, (long long)VPAD * DD);

  // ---- embedding ----
  embed_kernel<<<blks((long long)MTOT * DD), 256, 0, stream>>>(x, tok, pos, h);

  // ---- transformer layers ----
  for (int l = 0; l < LL; ++l) {
    // attn block
    ln_kernel<<<MTOT, 256, 0, stream>>>(h, ln1_s + l * DD, ln1_b + l * DD, xb);
    gemm_wmma_bf16<2,false,false><<<dim3(3 * DD / 128, MTOT / 64, 1), 256, 0, stream>>>(
        xb, qkvwt + (size_t)l * 3 * DD * DD, qkv_b + (size_t)l * 3 * DD,
        nullptr, qkvf, DD, 3 * DD, 3 * DD, 0, 0, 1, 0, 0);
    split_qkv_kernel<<<blks((long long)MTOT * 3 * DD), 256, 0, stream>>>(qkvf, qb, kb, vT);
    // scores = q @ k^T   (per b,h)
    gemm_wmma_bf16<2,false,false><<<dim3(TT / 128, TT / 64, BB * HH), 256, 0, stream>>>(
        qb, kb, nullptr, nullptr, scores, HDIM, TT, TT,
        (long long)TT * HDIM, (long long)TT * HDIM, 1, (long long)TT * TT, 0);
    softmax_kernel<<<BB * HH * TT, 256, 0, stream>>>(scores, attn);
    // o = attn @ v  -> bf16 stored directly in merged [B,T,D] layout
    gemm_wmma_bf16<1,false,true><<<dim3(HDIM / 64, TT / 64, BB * HH), 256, 0, stream>>>(
        attn, vT, nullptr, nullptr, obf, TT, HDIM, DD,
        (long long)TT * TT, (long long)TT * HDIM,
        HH, (long long)TT * DD, (long long)HDIM);
    // h = h + o @ out_w + out_b
    gemm_wmma_bf16<2,false,false><<<dim3(DD / 128, MTOT / 64, 1), 256, 0, stream>>>(
        obf, outwt + (size_t)l * DD * DD, out_b + (size_t)l * DD,
        h, h, DD, DD, DD, 0, 0, 1, 0, 0);

    // mlp block: ffb = gelu(ln2(h) @ ff1_w + b)  (bf16 out), h += ffb @ ff2_w + b
    ln_kernel<<<MTOT, 256, 0, stream>>>(h, ln2_s + l * DD, ln2_b + l * DD, xb);
    gemm_wmma_bf16<2,true,true><<<dim3(FF / 128, MTOT / 64, 1), 256, 0, stream>>>(
        xb, ff1wt + (size_t)l * FF * DD, ff1_b + (size_t)l * FF,
        nullptr, ffb, DD, FF, FF, 0, 0, 1, 0, 0);
    gemm_wmma_bf16<2,false,false><<<dim3(DD / 128, MTOT / 64, 1), 256, 0, stream>>>(
        ffb, ff2wt + (size_t)l * DD * FF, ff2_b + (size_t)l * DD,
        h, h, FF, DD, DD, 0, 0, 1, 0, 0);
  }

  // ---- final LN + head ----
  ln_kernel<<<MTOT, 256, 0, stream>>>(h, lnf_s, lnf_b, xb);
  gemm_wmma_bf16<2,false,false><<<dim3(VPAD / 128, MTOT / 64, 1), 256, 0, stream>>>(
      xb, headwt, head_b, nullptr, logits, DD, VV, VV, 0, 0, 1, 0, 0);
}
